// Quantize_9234179687622
// MI455X (gfx1250) — matrix operand order repaired and checked by reading, not compiled, not run on hardware
//
#include <hip/hip_runtime.h>
#include <hip/hip_bf16.h>

typedef __attribute__((ext_vector_type(16))) _Float16 v16h;
typedef __attribute__((ext_vector_type(8)))  float    v8f;

#define CDIM    256
#define NEMB    1024
#define Q_ELEMS (16u*256u*64u*64u)   /* 16777216 quantize elements */

// ---------------------------------------------------------------------------
// Kernel 1: embT[e][k] = (f16)embed[k][e]  (WMMA-B friendly, contiguous K),
//           enorm[e]   = sum_k embed[k][e]^2 in exact fp32.
// ---------------------------------------------------------------------------
__global__ __launch_bounds__(256)
void vq_prep(const float* __restrict__ embed,
             _Float16* __restrict__ embT,
             float* __restrict__ enorm) {
  const int e = blockIdx.x;     // 0..1023
  const int k = threadIdx.x;    // 0..255
  const float v = embed[(size_t)k * NEMB + e];
  embT[(size_t)e * CDIM + k] = (_Float16)v;

  float sq = v * v;
  #pragma unroll
  for (int off = 16; off; off >>= 1) sq += __shfl_xor(sq, off, 32);
  __shared__ float w[8];
  if ((threadIdx.x & 31) == 0) w[threadIdx.x >> 5] = sq;
  __syncthreads();
  if (threadIdx.x == 0) {
    float s = 0.f;
    #pragma unroll
    for (int i = 0; i < 8; ++i) s += w[i];
    enorm[e] = s;
  }
}

// ---------------------------------------------------------------------------
// Kernel 2: one wave32 = 16 consecutive pixels (one M-tile).
//   A tile (16x256) lives in 8 v16h fragments (registers), reused for all
//   64 code tiles.  dot = v_wmma_f32_16x16x32_f16; dist = ||e||^2 - 2*dot.
// ---------------------------------------------------------------------------
__global__ __launch_bounds__(256)
void vq_main(const float* __restrict__ x,
             const float* __restrict__ embed,
             const _Float16* __restrict__ embT,
             const float* __restrict__ enorm,
             float* __restrict__ quant,
             float* __restrict__ idx_out,
             float* __restrict__ partials) {
  __shared__ int   sidx[128];
  __shared__ float wred[8];

  const int lane = threadIdx.x & 31;
  const int wave = threadIdx.x >> 5;
  const int m    = lane & 15;      // column-within-tile / pixel-within-tile
  const int hi   = lane >> 4;      // lane half (ISA K-split)

  const int n0 = blockIdx.x * 128 + wave * 16;   // first row of this wave
  const int b  = n0 >> 12;                       // image (4096 pixels each)
  const int p  = (n0 & 4095) + m;                // pixel within image

  // ---- load A (16 rows x 256 feats) -> f16 fragments, contiguous K layout:
  //      lanes 0-15:  h0..15 -> K 0..15
  //      lanes 16-31: h0..15 -> K 16..31   (per 32-K tile)
  const float* xb = x + (size_t)b * (CDIM * 4096) + p;
  v16h afrag[8];
  float rowsq = 0.f;                 // this lane's half of ||x_row||^2 (fp32)
  #pragma unroll
  for (int kt = 0; kt < 8; ++kt) {
    v16h a;
    #pragma unroll
    for (int h = 0; h < 16; ++h) {
      const int kk = kt * 32 + hi * 16 + h;
      const float v = xb[(size_t)kk * 4096];
      rowsq += v * v;
      a[h] = (_Float16)v;
    }
    afrag[kt] = a;
  }

  // ---- sweep 1024 codes, 16 per tile; track per-slot argmin --------------
  float bestd[8];
  int   besti[8];
  #pragma unroll
  for (int r = 0; r < 8; ++r) { bestd[r] = 3.4e38f; besti[r] = 0; }

  #pragma unroll 1
  for (int nt = 0; nt < 64; ++nt) {
    const int ncol = nt * 16 + m;
    // B (32x16) fragment: lane holds 16 contiguous K for its column:
    //   lanes 0-15 -> K kt*32+0..15, lanes 16-31 -> K kt*32+16..31
    const _Float16* bp = embT + (size_t)ncol * CDIM + hi * 16;
    v8f c = {0.f, 0.f, 0.f, 0.f, 0.f, 0.f, 0.f, 0.f};
    #pragma unroll
    for (int kt = 0; kt < 8; ++kt) {
      const v16h bf = *(const v16h*)(bp + kt * 32);
      c = __builtin_amdgcn_wmma_f32_16x16x32_f16(false, afrag[kt], false, bf,
                                                 (short)0, c, false, false);
    }
    const float en = enorm[ncol];
    #pragma unroll
    for (int r = 0; r < 8; ++r) {          // row = hi*8 + r (C/D layout)
      const float d = en - 2.0f * c[r];
      if (d < bestd[r]) { bestd[r] = d; besti[r] = ncol; }
    }
  }

  // ---- min-reduce across the 16 lanes (columns) of each half -------------
  #pragma unroll
  for (int off = 8; off; off >>= 1) {
    #pragma unroll
    for (int r = 0; r < 8; ++r) {
      const float od = __shfl_xor(bestd[r], off, 16);
      const int   oi = __shfl_xor(besti[r], off, 16);
      if (od < bestd[r] || (od == bestd[r] && oi < besti[r])) {
        bestd[r] = od; besti[r] = oi;
      }
    }
  }

  // lanes 0 / 16 own rows 0..7 / 8..15
  if (m == 0) {
    #pragma unroll
    for (int r = 0; r < 8; ++r) {
      const int row = hi * 8 + r;
      sidx[wave * 16 + row] = besti[r];
      idx_out[n0 + row] = (float)besti[r];
    }
  }

  // ---- diff: sum over rows of (||x||^2 + bestd), fixed-tree reduction ----
  float t = rowsq;
  if (m == 0) {
    float s = 0.f;
    #pragma unroll
    for (int r = 0; r < 8; ++r) s += bestd[r];
    t += s;
  }
  #pragma unroll
  for (int off = 16; off; off >>= 1) t += __shfl_xor(t, off, 32);
  if (lane == 0) wred[wave] = t;
  __syncthreads();
  if (threadIdx.x == 0) {
    float s = 0.f;
    #pragma unroll
    for (int i = 0; i < 8; ++i) s += wred[i];
    partials[blockIdx.x] = s * (1.0f / (float)Q_ELEMS);
  }

  // ---- gather winning codes (exact fp32) and write coalesced along W ----
  const int myidx = sidx[wave * 16 + m];       // idx for this lane's pixel
  float* ob = quant + (size_t)b * (CDIM * 4096) + p;
  #pragma unroll 4
  for (int c = hi; c < CDIM; c += 2) {         // 2 channels / iteration
    ob[(size_t)c * 4096] = embed[(size_t)c * NEMB + myidx];
  }
}

// ---------------------------------------------------------------------------
// Kernel 3: deterministic sum of 512 block partials -> diff scalar.
// ---------------------------------------------------------------------------
__global__ __launch_bounds__(256)
void vq_final(const float* __restrict__ partials, float* __restrict__ diff_out) {
  float s = partials[threadIdx.x] + partials[threadIdx.x + 256];
  #pragma unroll
  for (int off = 16; off; off >>= 1) s += __shfl_xor(s, off, 32);
  __shared__ float w[8];
  if ((threadIdx.x & 31) == 0) w[threadIdx.x >> 5] = s;
  __syncthreads();
  if (threadIdx.x == 0) {
    float t = 0.f;
    #pragma unroll
    for (int i = 0; i < 8; ++i) t += w[i];
    *diff_out = t;
  }
}

// ---------------------------------------------------------------------------
extern "C" void kernel_launch(void* const* d_in, const int* in_sizes, int n_in,
                              void* d_out, int out_size, void* d_ws, size_t ws_size,
                              hipStream_t stream) {
  const float* x     = (const float*)d_in[0];   // [16,256,64,64]
  const float* embed = (const float*)d_in[1];   // [256,1024]

  float* out   = (float*)d_out;
  float* quant = out;                 // 16,777,216 f32
  float* diff  = out + Q_ELEMS;       // 1 f32
  float* idxo  = out + Q_ELEMS + 1;   // 65,536 f32 (indices as float)

  char*     ws       = (char*)d_ws;
  float*    enorm    = (float*)ws;             // 1024 f32   (4 KB)
  float*    partials = (float*)(ws + 4096);    // 512 f32    (2 KB)
  _Float16* embT     = (_Float16*)(ws + 8192); // 1024x256 f16 (512 KB)

  vq_prep <<<1024, 256, 0, stream>>>(embed, embT, enorm);
  vq_main <<< 512, 256, 0, stream>>>(x, embed, embT, enorm, quant, idxo, partials);
  vq_final<<<   1, 256, 0, stream>>>(partials, diff);
}
